// RGCNEncoder_41412074668232
// MI455X (gfx1250) — compile-verified, matrix operand order checked by reading, not run on hardware
//
#include <hip/hip_runtime.h>

// ---------------------------------------------------------------------------
// RGCN encoder for MI455X (gfx1250, wave32, WMMA)
//   layer: h_out = mean_agg(h)[r] @ W_r (sum over r) + h @ root + b
//   BN (biased var) + ELU between the two layers.
// Aggregate-first formulation keeps the 51.2MB feature table L2-resident
// across all 4 relation scatter passes (192MB L2).
// GEMM: bf16 WMMA, weight matrix pre-swizzled in LDS into B-fragment order
// so every fragment load is a contiguous 32B ds_load_b128 pair.
// ---------------------------------------------------------------------------

typedef __attribute__((ext_vector_type(16))) __bf16 v16bf;
typedef __attribute__((ext_vector_type(8)))  float  v8f;

#define DIM 128          // IN == HID == OUT == 128
#define NREL 4

__device__ __forceinline__ unsigned short f2bf(float f) {
    unsigned int u = __builtin_bit_cast(unsigned int, f);
    u += 0x7fffu + ((u >> 16) & 1u);          // round-to-nearest-even
    return (unsigned short)(u >> 16);
}
__device__ __forceinline__ __bf16 bfbits(unsigned short s) {
    return __builtin_bit_cast(__bf16, s);
}

// --------------------------- utility kernels -------------------------------

__global__ void zero_f32_kernel(float* __restrict__ p, int n) {
    int i = blockIdx.x * blockDim.x + threadIdx.x;
    if (i < n) p[i] = 0.0f;
}

__global__ void count_kernel(const int* __restrict__ dst, int* __restrict__ cnt, int nE) {
    int i = blockIdx.x * blockDim.x + threadIdx.x;
    if (i < nE) atomicAdd(&cnt[dst[i]], 1);
}

__global__ void inv_kernel(const int* __restrict__ cnt, float* __restrict__ inv, int n) {
    int i = blockIdx.x * blockDim.x + threadIdx.x;
    if (i < n) inv[i] = 1.0f / fmaxf((float)cnt[i], 1.0f);
}

// ----------------------------- edge scatter --------------------------------
// One wave32 per edge: agg[dst] += h[src] * inv[dst]   (mean folded in).
__global__ void scatter_kernel(const float* __restrict__ h,
                               const int* __restrict__ src,
                               const int* __restrict__ dst,
                               const float* __restrict__ invr,
                               float* __restrict__ agg, int nE) {
    int gid  = blockIdx.x * blockDim.x + threadIdx.x;
    int e    = gid >> 5;
    int lane = gid & 31;
    if (e >= nE) return;
    int s = src[e];
    int d = dst[e];
    float w = invr[d];
    const float4 v = ((const float4*)(h + (size_t)s * DIM))[lane];
    float* out = agg + (size_t)d * DIM + lane * 4;
    unsafeAtomicAdd(out + 0, v.x * w);
    unsafeAtomicAdd(out + 1, v.y * w);
    unsafeAtomicAdd(out + 2, v.z * w);
    unsafeAtomicAdd(out + 3, v.w * w);
}

// ------------------------------- GEMM (WMMA) -------------------------------
// Out[tile 128 x 128] (+)= H[tile 128 x 128] @ W[128 x 128]
// mode 0: Out = H@W + bias(col)     mode 1: Out += H@W (C loaded from Out)
// 8 waves/block, each wave: 16-row strip, 8 C tiles of 16x16, K = 4 x 32.
//
// ISA fragment mapping (wave32, v_wmma_f32_16x16x32_bf16):
//   A (16x32): lane<16 -> row=lane,     K = {0..7} u {16..23}
//              lane>=16 -> row=lane-16, K = {8..15} u {24..31}   (2 bf16/VGPR)
//   B (32x16): same K pattern per lane, col = lane%16 (assumed symmetric)
//   C/D:       VGPR j -> row j (lanes 0-15) / j+8 (lanes 16-31), col = lane%16
__global__ __launch_bounds__(256)
void gemm_kernel(const float* __restrict__ H, const float* __restrict__ W,
                 const float* __restrict__ bias, float* __restrict__ Out,
                 int n, int mode) {
    __shared__ unsigned short sA[128][128];     // bf16 activations, row-major (32 KB)
    __shared__ unsigned short sBsw[32 * 32 * 16]; // bf16 weights, B-fragment order (32 KB)
    // sBsw layout: fragment (kk,t) -> 32 lanes x 16 bf16 contiguous:
    //   sBsw[(((kk*8 + t) * 32) + lane) * 16 + e]

    const int tid   = threadIdx.x;
    const int tileM = blockIdx.x * 128;

    // Stage H tile (row-major bf16) + W (fragment-swizzled bf16) into LDS.
#pragma unroll
    for (int it = 0; it < 16; ++it) {
        int idx  = it * 256 + tid;        // 0..4095, one float4 each
        int row  = idx >> 5;              // 0..127
        int col4 = (idx & 31) * 4;
        int grow = tileM + row;
        float4 v = make_float4(0.f, 0.f, 0.f, 0.f);
        if (grow < n) v = *(const float4*)(H + (size_t)grow * DIM + col4);
        unsigned long long pk =
            (unsigned long long)f2bf(v.x) |
            ((unsigned long long)f2bf(v.y) << 16) |
            ((unsigned long long)f2bf(v.z) << 32) |
            ((unsigned long long)f2bf(v.w) << 48);
        *(unsigned long long*)&sA[row][col4] = pk;   // one 8B LDS store

        // W element (k=row, c=col4+i) -> fragment-swizzled position
        float4 wv = *(const float4*)(W + (size_t)row * DIM + col4);
        int k    = row;
        int kk   = k >> 5;
        int kin  = k & 31;
        int e    = ((kin >> 4) & 1) * 8 + (kin & 7);   // element within fragment
        int hi   = (kin >> 3) & 1;                     // lane group select
        float wf[4] = {wv.x, wv.y, wv.z, wv.w};
#pragma unroll
        for (int i = 0; i < 4; ++i) {
            int c    = col4 + i;
            int t    = c >> 4;
            int lane = hi * 16 + (c & 15);
            sBsw[(((kk * 8 + t) * 32) + lane) * 16 + e] = f2bf(wf[i]);
        }
    }
    __syncthreads();

    const int lane = tid & 31;
    const int wv_  = tid >> 5;            // wave id 0..7
    const int mrow = wv_ * 16;
    const int rA   = mrow + (lane & 15);  // A row for this lane
    const int kb   = ((lane >> 4) & 1) * 8;
    const int c15  = lane & 15;
    const int rofs = ((lane >> 4) & 1) * 8;

    v8f acc[8];
    if (mode == 0) {
#pragma unroll
        for (int t = 0; t < 8; ++t) {
            float bv = bias[t * 16 + c15];
#pragma unroll
            for (int j = 0; j < 8; ++j) acc[t][j] = bv;
        }
    } else {
#pragma unroll
        for (int t = 0; t < 8; ++t) {
#pragma unroll
            for (int j = 0; j < 8; ++j) {
                int row = tileM + mrow + j + rofs;
                acc[t][j] = (row < n) ? Out[(size_t)row * DIM + t * 16 + c15] : 0.f;
            }
        }
    }

#pragma unroll
    for (int kk = 0; kk < 4; ++kk) {
        // A fragment: two contiguous 16B chunks -> 2x ds_load_b128
        v16bf a;
#pragma unroll
        for (int e = 0; e < 16; ++e)
            a[e] = bfbits(sA[rA][kk * 32 + (e >> 3) * 16 + kb + (e & 7)]);
#pragma unroll
        for (int t = 0; t < 8; ++t) {
            // B fragment: 32B contiguous per lane -> 2x ds_load_b128
            const v16bf b = *(const v16bf*)&sBsw[(((kk * 8 + t) * 32) + lane) * 16];
            acc[t] = __builtin_amdgcn_wmma_f32_16x16x32_bf16(
                false, a, false, b, (short)0, acc[t], false, false);
        }
    }

#pragma unroll
    for (int t = 0; t < 8; ++t) {
#pragma unroll
        for (int j = 0; j < 8; ++j) {
            int row = tileM + mrow + j + rofs;
            if (row < n) Out[(size_t)row * DIM + t * 16 + c15] = acc[t][j];
        }
    }
}

// ------------------------------ BatchNorm + ELU ----------------------------

__global__ void bn_stats_kernel(const float* __restrict__ h,
                                float* __restrict__ sum, float* __restrict__ sq,
                                int n) {
    int c = threadIdx.x;   // 128 threads, one column each
    float s = 0.f, q = 0.f;
    for (int row = blockIdx.x; row < n; row += gridDim.x) {
        float x = h[(size_t)row * DIM + c];
        s += x;
        q += x * x;
    }
    unsafeAtomicAdd(&sum[c], s);
    unsafeAtomicAdd(&sq[c], q);
}

__global__ void bn_final_kernel(const float* __restrict__ sum, const float* __restrict__ sq,
                                const float* __restrict__ gamma, const float* __restrict__ beta,
                                float* __restrict__ scale, float* __restrict__ shift, int n) {
    int c = threadIdx.x;
    float m  = sum[c] / (float)n;
    float v  = sq[c] / (float)n - m * m;          // biased variance
    float rs = rsqrtf(v + 1e-5f);
    float sc = rs * gamma[c];
    scale[c] = sc;
    shift[c] = beta[c] - m * sc;
}

__global__ void bn_apply_elu_kernel(float* __restrict__ h,
                                    const float* __restrict__ scale,
                                    const float* __restrict__ shift, int n4) {
    int i = blockIdx.x * blockDim.x + threadIdx.x;
    if (i >= n4) return;
    float4 v = ((float4*)h)[i];
    int cb = (i & 31) * 4;
    v.x = v.x * scale[cb + 0] + shift[cb + 0];
    v.y = v.y * scale[cb + 1] + shift[cb + 1];
    v.z = v.z * scale[cb + 2] + shift[cb + 2];
    v.w = v.w * scale[cb + 3] + shift[cb + 3];
    v.x = v.x > 0.f ? v.x : (expf(v.x) - 1.f);
    v.y = v.y > 0.f ? v.y : (expf(v.y) - 1.f);
    v.z = v.z > 0.f ? v.z : (expf(v.z) - 1.f);
    v.w = v.w > 0.f ? v.w : (expf(v.w) - 1.f);
    ((float4*)h)[i] = v;
}

// --------------------------------- host ------------------------------------

static inline char* align512(char* p) {
    return (char*)(((uintptr_t)p + 511) & ~(uintptr_t)511);
}

extern "C" void kernel_launch(void* const* d_in, const int* in_sizes, int n_in,
                              void* d_out, int out_size, void* d_ws, size_t ws_size,
                              hipStream_t stream) {
    const int* edges[NREL] = {(const int*)d_in[0], (const int*)d_in[1],
                              (const int*)d_in[2], (const int*)d_in[3]};
    const float* emb   = (const float*)d_in[4];
    const float* W0    = (const float*)d_in[5];
    const float* root0 = (const float*)d_in[6];
    const float* b0    = (const float*)d_in[7];
    const float* gamma = (const float*)d_in[8];
    const float* beta  = (const float*)d_in[9];
    const float* W1    = (const float*)d_in[10];
    const float* root1 = (const float*)d_in[11];
    const float* b1    = (const float*)d_in[12];
    float* out = (float*)d_out;

    const int E = in_sizes[0] / 2;
    const int N = in_sizes[4] / DIM;

    // workspace carve
    char* p = (char*)d_ws;
    int*   cnt  = (int*)p;            p = align512(p + (size_t)NREL * N * sizeof(int));
    float* inv  = (float*)p;          p = align512(p + (size_t)NREL * N * sizeof(float));
    float* agg  = (float*)p;          p = align512(p + (size_t)N * DIM * sizeof(float));
    float* htmp = (float*)p;          p = align512(p + (size_t)N * DIM * sizeof(float));
    float* bnsum   = (float*)p;       p = align512(p + 128 * sizeof(float));
    float* bnsq    = (float*)p;       p = align512(p + 128 * sizeof(float));
    float* bnscale = (float*)p;       p = align512(p + 128 * sizeof(float));
    float* bnshift = (float*)p;       p = align512(p + 128 * sizeof(float));
    (void)ws_size; (void)n_in; (void)out_size;

    const int T = 256;
    const int featN   = N * DIM;
    const int gB      = (N + 127) / 128;                 // gemm blocks
    const int zB_feat = (featN + T - 1) / T;
    const int zB_cnt  = (NREL * N + T - 1) / T;
    const int cB      = (E + T - 1) / T;
    const int sB      = (E * 32 + T - 1) / T;            // scatter: 1 wave/edge
    const int aB      = (featN / 4 + T - 1) / T;         // bn apply (float4)

    // in-degree counts per (relation, node) -> reciprocal (shared by both layers)
    zero_f32_kernel<<<zB_cnt, T, 0, stream>>>((float*)cnt, NREL * N);
    for (int r = 0; r < NREL; ++r)
        count_kernel<<<cB, T, 0, stream>>>(edges[r] + E, cnt + (size_t)r * N, E);
    inv_kernel<<<zB_cnt, T, 0, stream>>>(cnt, inv, NREL * N);

    // ---------------- layer 0 ----------------
    gemm_kernel<<<gB, T, 0, stream>>>(emb, root0, b0, htmp, N, 0);
    for (int r = 0; r < NREL; ++r) {
        zero_f32_kernel<<<zB_feat, T, 0, stream>>>(agg, featN);
        scatter_kernel<<<sB, T, 0, stream>>>(emb, edges[r], edges[r] + E,
                                             inv + (size_t)r * N, agg, E);
        gemm_kernel<<<gB, T, 0, stream>>>(agg, W0 + (size_t)r * DIM * DIM,
                                          nullptr, htmp, N, 1);
    }

    // BatchNorm (training stats, biased var) + ELU, in place on htmp
    zero_f32_kernel<<<1, T, 0, stream>>>(bnsum, 128);
    zero_f32_kernel<<<1, T, 0, stream>>>(bnsq, 128);
    bn_stats_kernel<<<256, 128, 0, stream>>>(htmp, bnsum, bnsq, N);
    bn_final_kernel<<<1, 128, 0, stream>>>(bnsum, bnsq, gamma, beta, bnscale, bnshift, N);
    bn_apply_elu_kernel<<<aB, T, 0, stream>>>(htmp, bnscale, bnshift, featN / 4);

    // ---------------- layer 1 ----------------
    gemm_kernel<<<gB, T, 0, stream>>>(htmp, root1, b1, out, N, 0);
    for (int r = 0; r < NREL; ++r) {
        zero_f32_kernel<<<zB_feat, T, 0, stream>>>(agg, featN);
        scatter_kernel<<<sB, T, 0, stream>>>(htmp, edges[r], edges[r] + E,
                                             inv + (size_t)r * N, agg, E);
        gemm_kernel<<<gB, T, 0, stream>>>(agg, W1 + (size_t)r * DIM * DIM,
                                          nullptr, out, N, 1);
    }
}
